// BiGRU_82437602279873
// MI455X (gfx1250) — matrix operand order repaired and checked by reading, not compiled, not run on hardware
//
#include <hip/hip_runtime.h>
#include <hip/hip_bf16.h>

// ---------------------------------------------------------------------------
// Problem constants
// ---------------------------------------------------------------------------
#define BB   32
#define TT   1024
#define HH   768
#define G3H  2304          // 3*H
#define KK   768           // reduction dim (DIN == H == 768)
#define MTOT (BB*TT)       // 32768 rows for the input-projection GEMM

typedef __attribute__((ext_vector_type(8)))  float          v8f;
typedef __attribute__((ext_vector_type(16))) __bf16         v16bf;
typedef __attribute__((ext_vector_type(8)))  unsigned short us8;

// ---------------------------------------------------------------------------
// bf16 helpers (store bf16 as ushort, RNE conversion)
// ---------------------------------------------------------------------------
__device__ __forceinline__ unsigned short f2bf(float f) {
    union { float f; unsigned u; } v; v.f = f;
    unsigned r = v.u + 0x7fffu + ((v.u >> 16) & 1u);
    return (unsigned short)(r >> 16);
}

// ---------------------------------------------------------------------------
// WMMA wrapper: D = A(16x32 bf16) * B(32x16 bf16) + C(f32)
// ---------------------------------------------------------------------------
__device__ __forceinline__ v8f wmma_bf16(v16bf a, v16bf b, v8f c) {
    return __builtin_amdgcn_wmma_f32_16x16x32_bf16(
        /*neg_a=*/false, a, /*neg_b=*/false, b,
        /*c_mod=*/(short)0, c, /*reuse_a=*/false, /*reuse_b=*/false);
}

// A fragment (16x32, row-major source, leading dim ld):
//   lane l: row = base + (l&15); K values = ka+0..7 and ka+16..23, ka=(l>>4)*8
__device__ __forceinline__ v16bf load_frag_A(const unsigned short* a,
                                             int rowbase, int k0, int ld, int lane) {
    int r  = rowbase + (lane & 15);
    int ka = k0 + ((lane >> 4) << 3);
    union { us8 u[2]; v16bf v; } x;
    x.u[0] = *(const us8*)(a + (size_t)r * ld + ka);
    x.u[1] = *(const us8*)(a + (size_t)r * ld + ka + 16);
    return x.v;
}

// B fragment (32x16). B(k,n) = W[nrow + n][k] with W row-major [N][K]:
//   lane l: col = nbase + (l&15); K values = kb+0..7 and kb+16..23, kb=(l>>4)*8
__device__ __forceinline__ v16bf load_frag_B(const unsigned short* w,
                                             int nbase, int k0, int ld, int lane) {
    int n  = nbase + (lane & 15);
    int kb = k0 + ((lane >> 4) << 3);
    union { us8 u[2]; v16bf v; } x;
    x.u[0] = *(const us8*)(w + (size_t)n * ld + kb);
    x.u[1] = *(const us8*)(w + (size_t)n * ld + kb + 16);
    return x.v;
}

// ---------------------------------------------------------------------------
// Kernel 1: f32 -> bf16 weight conversion (grid-stride)
// ---------------------------------------------------------------------------
__global__ void cvt_bf16_kernel(const float* __restrict__ src,
                                unsigned short* __restrict__ dst, int n) {
    for (int i = blockIdx.x * blockDim.x + threadIdx.x; i < n;
         i += gridDim.x * blockDim.x)
        dst[i] = f2bf(src[i]);
}

// ---------------------------------------------------------------------------
// Kernel 2: per-layer prep — build bf16 input seq (optionally time-flipped),
//           zero both h double-buffers and the barrier counter.
// ---------------------------------------------------------------------------
__global__ void prep_kernel(const float* __restrict__ src, int flip,
                            unsigned short* __restrict__ dst,
                            float* __restrict__ hf, unsigned short* __restrict__ hb,
                            unsigned int* __restrict__ ctr) {
    const int N = BB * TT * HH;
    for (int i = blockIdx.x * blockDim.x + threadIdx.x; i < N;
         i += gridDim.x * blockDim.x) {
        int b = i / (TT * HH);
        int rem = i - b * (TT * HH);
        int t = rem / HH;
        int c = rem - t * HH;
        int si = flip ? (b * TT + (TT - 1 - t)) * HH + c : i;
        dst[i] = f2bf(src[si]);
        if (i < 2 * BB * HH) { hf[i] = 0.0f; hb[i] = 0; }
        if (i == 0) *ctr = 0u;
    }
}

// ---------------------------------------------------------------------------
// Kernel 3: input projection GEMM  xg[M,3H] = in[M,K](bf16) * Wih[3H,K]^T + b_ih
// One wave per 32x32 output tile, 2x2 WMMA accumulators, 24 k-steps.
// ---------------------------------------------------------------------------
__global__ __launch_bounds__(256) void xg_gemm_kernel(
    const unsigned short* __restrict__ Abf,   // [MTOT][KK]
    const unsigned short* __restrict__ Wbf,   // [G3H][KK]
    const float* __restrict__ bias,           // [G3H]
    float* __restrict__ xg)                   // [MTOT][G3H]
{
    const int lane   = threadIdx.x & 31;
    const int waveId = blockIdx.x * 8 + (threadIdx.x >> 5);
    const int NT = G3H / 32;                  // 72
    int mt = waveId / NT, nt = waveId % NT;
    int mbase = mt * 32, nbase = nt * 32;

    v8f acc00 = {}, acc01 = {}, acc10 = {}, acc11 = {};
    for (int k0 = 0; k0 < KK; k0 += 32) {
        v16bf a0 = load_frag_A(Abf, mbase,      k0, KK, lane);
        v16bf a1 = load_frag_A(Abf, mbase + 16, k0, KK, lane);
        v16bf b0 = load_frag_B(Wbf, nbase,      k0, KK, lane);
        v16bf b1 = load_frag_B(Wbf, nbase + 16, k0, KK, lane);
        acc00 = wmma_bf16(a0, b0, acc00);
        acc01 = wmma_bf16(a0, b1, acc01);
        acc10 = wmma_bf16(a1, b0, acc10);
        acc11 = wmma_bf16(a1, b1, acc11);
    }
    // C/D layout: elem i -> row = i + (lane>=16)*8, col = lane&15
    int rb = (lane >> 4) << 3;
    int cl = lane & 15;
    #pragma unroll
    for (int i = 0; i < 8; ++i) {
        int r0 = mbase + rb + i, r1 = mbase + 16 + rb + i;
        int c0 = nbase + cl,     c1 = nbase + 16 + cl;
        xg[(size_t)r0 * G3H + c0] = acc00[i] + bias[c0];
        xg[(size_t)r0 * G3H + c1] = acc01[i] + bias[c1];
        xg[(size_t)r1 * G3H + c0] = acc10[i] + bias[c0];
        xg[(size_t)r1 * G3H + c1] = acc11[i] + bias[c1];
    }
}

// ---------------------------------------------------------------------------
// Kernel 4: persistent GRU recurrence for one layer.
// 48 blocks x 192 threads (6 waves). Block owns 16 H-columns; its W_hh slice
// (3 gates x 16 cols x 768 K, bf16 = 72KB) lives in LDS for all 1024 steps.
// Wave w: gate = w>>1, batch-half = w&1 -> one 16x16 WMMA tile of hg per step.
// h is double-buffered (bf16 for WMMA A + f32 for z*h) across the grid barrier.
// ---------------------------------------------------------------------------
__global__ __launch_bounds__(192, 1) void gru_rec_kernel(
    const unsigned short* __restrict__ whh,   // [G3H][KK] bf16
    const float* __restrict__ bhh,            // [G3H]
    const float* __restrict__ xg,             // [MTOT][G3H]
    const float* __restrict__ resid,          // prev layer out [B][T][H] or null
    float* __restrict__ out,                  // [B][T][H]
    float* __restrict__ hf,                   // [2][B][H] f32
    unsigned short* __restrict__ hb,          // [2][B][H] bf16
    unsigned int* __restrict__ ctr,
    int flip_out)
{
    __shared__ unsigned short lds_w[3 * 16 * KK];   // 73728 B
    __shared__ float lds_hg[3][BB][16];             //  6144 B

    const int tid  = threadIdx.x;
    const int lane = tid & 31;
    const int w    = tid >> 5;          // 0..5
    const int colbase = blockIdx.x * 16;

    // Preload W_hh slice: 3 contiguous 16x768 row blocks (rows g*768+colbase..+15)
    for (int g = 0; g < 3; ++g) {
        const uint4* s = (const uint4*)(whh + (size_t)(g * HH + colbase) * KK);
        uint4* d = (uint4*)(lds_w + g * 16 * KK);
        for (int i = tid; i < 16 * KK / 8; i += 192) d[i] = s[i];
    }
    __syncthreads();

    const int gate  = w >> 1;           // 0,0,1,1,2,2
    const int mbase = (w & 1) << 4;     // 0 / 16

    for (int t = 0; t < TT; ++t) {
        const unsigned short* hb_r = hb + (size_t)(t & 1) * BB * HH;
        unsigned short*       hb_w = hb + (size_t)((t + 1) & 1) * BB * HH;
        const float*          hf_r = hf + (size_t)(t & 1) * BB * HH;
        float*                hf_w = hf + (size_t)((t + 1) & 1) * BB * HH;

        // ---- hg tile: [16 batch rows] x [16 cols] over K=768 ----
        v8f acc = {};
        for (int k0 = 0; k0 < KK; k0 += 32) {
            v16bf a = load_frag_A(hb_r, mbase, k0, HH, lane);
            v16bf b = load_frag_B(lds_w, gate * 16, k0, KK, lane);
            acc = wmma_bf16(a, b, acc);
        }
        {
            int rb = (lane >> 4) << 3, cl = lane & 15;
            #pragma unroll
            for (int i = 0; i < 8; ++i)
                lds_hg[gate][mbase + rb + i][cl] = acc[i];
        }
        __syncthreads();

        // ---- gate math over this block's 32x16 elements ----
        for (int e = tid; e < BB * 16; e += 192) {
            int b  = e >> 4;
            int cl = e & 15;
            int c  = colbase + cl;
            float hr = lds_hg[0][b][cl] + bhh[c];
            float hz = lds_hg[1][b][cl] + bhh[HH + c];
            float hn = lds_hg[2][b][cl] + bhh[2 * HH + c];
            const float* xgp = xg + ((size_t)b * TT + t) * G3H;
            float xr = xgp[c], xz = xgp[HH + c], xn = xgp[2 * HH + c];
            float r = 1.0f / (1.0f + __expf(-(xr + hr)));
            float z = 1.0f / (1.0f + __expf(-(xz + hz)));
            float n = tanhf(xn + r * hn);
            float hprev = hf_r[b * HH + c];
            float hnew  = (1.0f - z) * n + z * hprev;
            hf_w[b * HH + c] = hnew;
            hb_w[b * HH + c] = f2bf(hnew);
            int tout = flip_out ? (TT - 1 - t) : t;
            float res = 0.0f;
            if (resid) res = resid[((size_t)b * TT + (TT - 1 - t)) * HH + c];
            out[((size_t)b * TT + tout) * HH + c] = hnew + res;
        }
        __syncthreads();

        // ---- grid-wide step barrier (monotonic counter, no reset race) ----
        if (tid == 0) {
            __threadfence();
            atomicAdd(ctr, 1u);
            unsigned target = 48u * (unsigned)(t + 1);
            while (atomicAdd(ctr, 0u) < target) __builtin_amdgcn_s_sleep(1);
        }
        __syncthreads();
    }
}

// ---------------------------------------------------------------------------
// Host side: workspace carve-up + per-layer pipeline
// ---------------------------------------------------------------------------
extern "C" void kernel_launch(void* const* d_in, const int* in_sizes, int n_in,
                              void* d_out, int out_size, void* d_ws, size_t ws_size,
                              hipStream_t stream) {
    (void)in_sizes; (void)n_in; (void)out_size; (void)ws_size;
    const float* x     = (const float*)d_in[0];
    const float* Wih0  = (const float*)d_in[1];
    const float* Whh0  = (const float*)d_in[2];
    const float* bih0  = (const float*)d_in[3];
    const float* bhh0  = (const float*)d_in[4];
    const float* Wihs  = (const float*)d_in[5];   // [3][2304][768]
    const float* Whhs  = (const float*)d_in[6];
    const float* bihs  = (const float*)d_in[7];   // [3][2304]
    const float* bhhs  = (const float*)d_in[8];

    const size_t WSZ = (size_t)G3H * KK;          // 1,769,472 elems per matrix
    char* p = (char*)d_ws;
    unsigned short* wih_bf = (unsigned short*)p;  p += 4 * WSZ * 2;
    unsigned short* whh_bf = (unsigned short*)p;  p += 4 * WSZ * 2;
    unsigned short* in_bf  = (unsigned short*)p;  p += (size_t)MTOT * HH * 2;
    float* xg   = (float*)p;                      p += (size_t)MTOT * G3H * 4;
    float* outA = (float*)p;                      p += (size_t)MTOT * HH * 4;
    float* outB = (float*)p;                      p += (size_t)MTOT * HH * 4;
    float* hf   = (float*)p;                      p += 2 * (size_t)BB * HH * 4;
    unsigned short* hb = (unsigned short*)p;      p += 2 * (size_t)BB * HH * 2;
    unsigned int* ctr = (unsigned int*)p;

    // Weights -> bf16 (layer 0 then stacked layers 1..3, contiguous)
    cvt_bf16_kernel<<<2048, 256, 0, stream>>>(Wih0, wih_bf,            (int)WSZ);
    cvt_bf16_kernel<<<2048, 256, 0, stream>>>(Wihs, wih_bf + WSZ, (int)(3 * WSZ));
    cvt_bf16_kernel<<<2048, 256, 0, stream>>>(Whh0, whh_bf,            (int)WSZ);
    cvt_bf16_kernel<<<2048, 256, 0, stream>>>(Whhs, whh_bf + WSZ, (int)(3 * WSZ));

    const int GEMM_BLOCKS = (MTOT / 32) * (G3H / 32) / 8;   // 9216

    float* prevOut = nullptr;
    for (int l = 0; l < 4; ++l) {
        const float* src = (l == 0) ? x : prevOut;
        const float* bih = (l == 0) ? bih0 : bihs + (size_t)(l - 1) * G3H;
        const float* bhh = (l == 0) ? bhh0 : bhhs + (size_t)(l - 1) * G3H;
        float* curOut = (l == 3) ? (float*)d_out : ((l & 1) ? outB : outA);

        prep_kernel<<<8192, 256, 0, stream>>>(src, (l > 0) ? 1 : 0,
                                              in_bf, hf, hb, ctr);
        xg_gemm_kernel<<<GEMM_BLOCKS, 256, 0, stream>>>(
            in_bf, wih_bf + (size_t)l * WSZ, bih, xg);
        gru_rec_kernel<<<48, 192, 0, stream>>>(
            whh_bf + (size_t)l * WSZ, bhh, xg,
            (l == 0) ? nullptr : prevOut, curOut, hf, hb, ctr,
            (l == 3) ? 1 : 0);

        prevOut = curOut;
    }
}
